// ProjectedAdaptiveLogSoftmax_9646496546972
// MI455X (gfx1250) — matrix operand order repaired and checked by reading, not compiled, not run on hardware
//
#include <hip/hip_runtime.h>

typedef __attribute__((ext_vector_type(16))) __bf16 v16bf;
typedef __attribute__((ext_vector_type(8)))  float  v8f;

union BF16x16 { v16bf v; unsigned uu[8]; };

// Pack two f32 into two bf16 (round-half-up) in one dword.
static __device__ inline unsigned pk2bf(float lo, float hi) {
  union { float f; unsigned u; } a, b;
  a.f = lo; b.f = hi;
  return __builtin_amdgcn_perm(b.u + 0x8000u, a.u + 0x8000u, 0x07060302u);
}
static __device__ inline unsigned short f2bf(float f) {
  union { float f; unsigned u; } x; x.f = f;
  return (unsigned short)((x.u + 0x8000u) >> 16);
}

// ---------------------------------------------------------------------------
// C = A[4096,K] @ W[outdim,K]^T (+bias); A and W are PRE-CONVERTED bf16.
// REQUIRES K % 32 == 0. One wave owns a 64x64 tile; steady-state loop is
// 16x global_load_b128 + 16x v_wmma_f32_16x16x32_bf16 per K-chunk of 32.
// mode 0: store C as f32 (stride out_ld) AND bf16 (stride out_ldb).
// mode 1: per-row (max, sum(exp)) partials over this 64-col chunk.
// ---------------------------------------------------------------------------
__global__ __launch_bounds__(32)
void gemm_bf16(const unsigned short* __restrict__ A, int lda, int K,
               const unsigned short* __restrict__ W, const unsigned short* __restrict__ Wx, int wsplit,
               const float* __restrict__ bias, const float* __restrict__ biasx, int bsplit,
               int outdim, int mode,
               float* __restrict__ Cout, int out_ld,
               unsigned short* __restrict__ Cb, int out_ldb,
               float* __restrict__ pmax, float* __restrict__ psum, int nchunks)
{
  const int lane = threadIdx.x;        // 0..31
  const int hi   = lane >> 4;          // half-wave index
  const int ln   = lane & 15;
  const int nbase = blockIdx.x * 64;
  const int mbase = blockIdx.y * 64;

  const unsigned short* wrow[4];
  float bv[4];
  bool  cvalid[4];
#pragma unroll
  for (int ct = 0; ct < 4; ++ct) {
    int col = nbase + ct * 16 + ln;
    cvalid[ct] = (col < outdim);
    int cc = cvalid[ct] ? col : (outdim - 1);
    wrow[ct] = (cc < wsplit) ? (W + (size_t)cc * K)
                             : (Wx + (size_t)(cc - wsplit) * K);
    float b = 0.f;
    if (bias) b = (cc < bsplit) ? bias[cc] : biasx[cc - bsplit];
    bv[ct] = b;
  }
  const unsigned short* arow[4];
#pragma unroll
  for (int rt = 0; rt < 4; ++rt)
    arow[rt] = A + (size_t)(mbase + rt * 16 + ln) * lda;

  v8f c[4][4];
#pragma unroll
  for (int rt = 0; rt < 4; ++rt)
#pragma unroll
    for (int ct = 0; ct < 4; ++ct)
#pragma unroll
      for (int v = 0; v < 8; ++v) c[rt][ct][v] = 0.f;

  for (int kb = 0; kb < K; kb += 32) {
    // A 16x32 bf16 layout: lanes0-15 e0..7=K+0..7, e8..15=K+16..23; lanes16-31 +8.
    BF16x16 a[4];
#pragma unroll
    for (int rt = 0; rt < 4; ++rt) {
      uint4 q0 = *(const uint4*)(arow[rt] + kb + 8 * hi);
      uint4 q1 = *(const uint4*)(arow[rt] + kb + 16 + 8 * hi);
      a[rt].uu[0] = q0.x; a[rt].uu[1] = q0.y; a[rt].uu[2] = q0.z; a[rt].uu[3] = q0.w;
      a[rt].uu[4] = q1.x; a[rt].uu[5] = q1.y; a[rt].uu[6] = q1.z; a[rt].uu[7] = q1.w;
    }
    // B 32x16: lane = column; e0..7=K+8*hi+0..7, e8..15=K+16+8*hi+0..7 (matches A).
#pragma unroll
    for (int ct = 0; ct < 4; ++ct) {
      const unsigned short* bp = wrow[ct] + kb + 8 * hi;
      __builtin_prefetch(bp + 32, 0, 3);     // next K-chunk
      uint4 r0 = *(const uint4*)(bp);
      uint4 r1 = *(const uint4*)(bp + 16);
      BF16x16 b;
      b.uu[0] = r0.x; b.uu[1] = r0.y; b.uu[2] = r0.z; b.uu[3] = r0.w;
      b.uu[4] = r1.x; b.uu[5] = r1.y; b.uu[6] = r1.z; b.uu[7] = r1.w;
#pragma unroll
      for (int rt = 0; rt < 4; ++rt)
        c[rt][ct] = __builtin_amdgcn_wmma_f32_16x16x32_bf16(
            false, a[rt].v, false, b.v, (short)0, c[rt][ct], false, false);
    }
  }

  if (mode == 0) {
    // C/D layout: VGPR v, lanes0-15 -> (M=v, N=ln), lanes16-31 -> (M=v+8, N=ln)
#pragma unroll
    for (int rt = 0; rt < 4; ++rt)
#pragma unroll
      for (int ct = 0; ct < 4; ++ct) {
        if (!cvalid[ct]) continue;
        int col = nbase + ct * 16 + ln;
#pragma unroll
        for (int v = 0; v < 8; ++v) {
          int row = mbase + rt * 16 + v + 8 * hi;
          float val = c[rt][ct][v] + bv[ct];
          Cout[(size_t)row * out_ld + col] = val;
          Cb[(size_t)row * out_ldb + col] = f2bf(val);
        }
      }
    return;
  }

  // mode 1: per-row max & sum(exp) over this 64-column chunk
  __shared__ float red[64], reds[64];
  red[lane] = -__builtin_inff(); red[lane + 32] = -__builtin_inff();
  __syncthreads();

#pragma unroll
  for (int rt = 0; rt < 4; ++rt)
#pragma unroll
    for (int v = 0; v < 8; ++v) {
      float mx = -__builtin_inff();
#pragma unroll
      for (int ct = 0; ct < 4; ++ct)
        if (cvalid[ct]) mx = fmaxf(mx, c[rt][ct][v] + bv[ct]);
      for (int off = 1; off < 16; off <<= 1)           // reduce within 16-lane half
        mx = fmaxf(mx, __shfl_xor(mx, off, 32));
      if (ln == 0) red[rt * 16 + v + 8 * hi] = mx;     // row = rt*16 + v + 8*hi
    }
  __syncthreads();

#pragma unroll
  for (int rt = 0; rt < 4; ++rt)
#pragma unroll
    for (int v = 0; v < 8; ++v) {
      float m = red[rt * 16 + v + 8 * hi];
      float s = 0.f;
#pragma unroll
      for (int ct = 0; ct < 4; ++ct)
        if (cvalid[ct]) s += __expf(c[rt][ct][v] + bv[ct] - m);
      for (int off = 1; off < 16; off <<= 1)
        s += __shfl_xor(s, off, 32);
      if (ln == 0) reds[rt * 16 + v + 8 * hi] = s;
    }
  __syncthreads();

  for (int it = 0; it < 2; ++it) {
    int r = lane + it * 32;
    int grow = mbase + r;
    pmax[(size_t)grow * nchunks + blockIdx.x] = red[r];
    psum[(size_t)grow * nchunks + blockIdx.x] = reds[r];
  }
}

// Streaming f32 -> packed bf16 pairs (one dword out per thread).
__global__ void cvt_bf16(const float* __restrict__ src, unsigned* __restrict__ dst, int npairs)
{
  int i = blockIdx.x * blockDim.x + threadIdx.x;
  if (i >= npairs) return;
  dst[i] = pk2bf(src[2 * i], src[2 * i + 1]);
}

// pTb[j][k] = bf16(p[k][j])   (p is [1024][d])
__global__ void transpose_cvt_p(const float* __restrict__ p, unsigned short* __restrict__ pTb,
                                int K, int d)
{
  int e = blockIdx.x * blockDim.x + threadIdx.x;
  if (e >= K * d) return;
  int k = e / d, j = e % d;
  pTb[(size_t)j * K + k] = f2bf(p[e]);
}

// w3 [10000][16] f32 -> w3b [10000][32] bf16, zero-padded
__global__ void pad_w3_bf16(const float* __restrict__ w3, unsigned short* __restrict__ w3b)
{
  int e = blockIdx.x * blockDim.x + threadIdx.x;
  if (e >= 10000 * 32) return;
  int r = e >> 5, c = e & 31;
  w3b[e] = (c < 16) ? f2bf(w3[r * 16 + c]) : (unsigned short)0;
}

// zero bf16 columns 16..31 of proj3b [4096][32]
__global__ void zero_pad_proj3b(unsigned short* __restrict__ p)
{
  int e = blockIdx.x * blockDim.x + threadIdx.x;
  if (e >= 4096 * 16) return;
  int r = e >> 4, c = e & 15;
  p[r * 32 + 16 + c] = (unsigned short)0;
}

// Combine per-chunk (max, sumexp) partials into per-row logsumexp.
__global__ void lse_combine(const float* __restrict__ pmax, const float* __restrict__ psum,
                            float* __restrict__ lse, int nchunks, int rows)
{
  int r = blockIdx.x * blockDim.x + threadIdx.x;
  if (r >= rows) return;
  const float* pm = pmax + (size_t)r * nchunks;
  const float* ps = psum + (size_t)r * nchunks;
  float m = -__builtin_inff();
  for (int i = 0; i < nchunks; ++i) m = fmaxf(m, pm[i]);
  float s = 0.f;
  for (int i = 0; i < nchunks; ++i) s += ps[i] * __expf(pm[i] - m);
  lse[r] = m + logf(s);
}

static __device__ inline float dotw(const float* a, const float* b, int n, int lane)
{
  float s = 0.f;
  for (int i = lane; i < n; i += 32) s += a[i] * b[i];
  for (int off = 1; off < 32; off <<= 1) s += __shfl_xor(s, off, 32);
  return s;
}

// One wave per token: exact-f32 gathered logits, assemble -log p(target)
__global__ __launch_bounds__(32)
void gather_nll(const float* __restrict__ proj0, const float* __restrict__ proj1,
                const float* __restrict__ proj2, const float* __restrict__ proj3,
                const float* __restrict__ w0, const float* __restrict__ b0,
                const float* __restrict__ w1, const float* __restrict__ b1,
                const float* __restrict__ w2, const float* __restrict__ b2,
                const float* __restrict__ w3, const float* __restrict__ b3,
                const float* __restrict__ cw, const float* __restrict__ cbias,
                const int* __restrict__ target,
                const float* __restrict__ lse0, const float* __restrict__ lse1,
                const float* __restrict__ lse2, const float* __restrict__ lse3,
                float* __restrict__ nll)
{
  int r = blockIdx.x;
  int lane = threadIdx.x;
  int t = target[r];
  float lp;
  if (t < 10000) {
    float g = dotw(proj0 + (size_t)r * 1024, w0 + (size_t)t * 1024, 1024, lane) + b0[t];
    lp = g - lse0[r];
  } else {
    int i, l, d; const float *P, *W, *B, *LSE;
    if (t < 20000)      { i = 1; l = 10000; d = 256; P = proj1; W = w1; B = b1; LSE = lse1; }
    else if (t < 40000) { i = 2; l = 20000; d = 64;  P = proj2; W = w2; B = b2; LSE = lse2; }
    else                { i = 3; l = 40000; d = 16;  P = proj3; W = w3; B = b3; LSE = lse3; }
    int cl = 3 - i;     // head column (10003 - i) == cluster_weight row (3 - i)
    float gC = dotw(proj0 + (size_t)r * 1024, cw + (size_t)cl * 1024, 1024, lane) + cbias[cl];
    int idx = t - l;
    float gt = dotw(P + (size_t)r * d, W + (size_t)idx * d, d, lane) + B[idx];
    lp = (gC - lse0[r]) + (gt - LSE[r]);
  }
  if (lane == 0) nll[r] = -lp;
}

__global__ __launch_bounds__(256)
void final_mean(const float* __restrict__ nll, float* __restrict__ out, int n)
{
  __shared__ float s[256];
  int tid = threadIdx.x;
  float acc = 0.f;
  for (int i = tid; i < n; i += 256) acc += nll[i];
  s[tid] = acc; __syncthreads();
  for (int off = 128; off > 0; off >>= 1) {
    if (tid < off) s[tid] += s[tid + off];
    __syncthreads();
  }
  if (tid == 0) out[0] = s[0] / (float)n;
}

extern "C" void kernel_launch(void* const* d_in, const int* in_sizes, int n_in,
                              void* d_out, int out_size, void* d_ws, size_t ws_size,
                              hipStream_t stream)
{
  const float* hidden = (const float*)d_in[0];
  const int*   target = (const int*)  d_in[1];
  const float* w0 = (const float*)d_in[2];  const float* b0 = (const float*)d_in[3];
  const float* p0 = (const float*)d_in[4];
  const float* w1 = (const float*)d_in[5];  const float* b1 = (const float*)d_in[6];
  const float* p1 = (const float*)d_in[7];
  const float* w2 = (const float*)d_in[8];  const float* b2 = (const float*)d_in[9];
  const float* p2 = (const float*)d_in[10];
  const float* w3 = (const float*)d_in[11]; const float* b3 = (const float*)d_in[12];
  const float* p3 = (const float*)d_in[13];
  const float* cw = (const float*)d_in[14]; const float* cb = (const float*)d_in[15];
  float* out = (float*)d_out;

  const int N = 4096, DP = 1024;
  const int dims[4]  = {1024, 256, 64, 16};
  const int HEAD = 10003;
  const int nch0 = (HEAD + 63) / 64;                 // 157
  const int nch[4] = {0, (10000 + 63) / 64, (20000 + 63) / 64, (10000 + 63) / 64};

  float* ws = (float*)d_ws;
  size_t off = 0;
  // bf16 buffers carved in float units (2 bf16 per float); all 16B aligned.
  unsigned short* hb  = (unsigned short*)(ws + off); off += (size_t)N * DP / 2;
  unsigned short* w0b = (unsigned short*)(ws + off); off += (size_t)10000 * 1024 / 2;
  unsigned short* cwb = (unsigned short*)(ws + off); off += (size_t)3 * 1024 / 2;
  unsigned short* w1b = (unsigned short*)(ws + off); off += (size_t)10000 * 256 / 2;
  unsigned short* w2b = (unsigned short*)(ws + off); off += (size_t)20000 * 64 / 2;
  unsigned short* w3b = (unsigned short*)(ws + off); off += (size_t)10000 * 32 / 2;
  unsigned short* pTb[4];
  for (int i = 0; i < 4; ++i) { pTb[i] = (unsigned short*)(ws + off); off += (size_t)dims[i] * DP / 2; }
  unsigned short* projb[4];
  const int pbld[4] = {1024, 256, 64, 32};           // bf16 proj strides (tail3 padded)
  for (int i = 0; i < 4; ++i) { projb[i] = (unsigned short*)(ws + off); off += (size_t)N * pbld[i] / 2; }
  float* proj0 = ws + off; off += (size_t)N * 1024;
  float* proj1 = ws + off; off += (size_t)N * 256;
  float* proj2 = ws + off; off += (size_t)N * 64;
  float* proj3 = ws + off; off += (size_t)N * 16;
  float* pmax0 = ws + off; off += (size_t)N * nch0;
  float* psum0 = ws + off; off += (size_t)N * nch0;
  float* pmaxT[4]; float* psumT[4];
  for (int i = 1; i < 4; ++i) {
    pmaxT[i] = ws + off; off += (size_t)N * nch[i];
    psumT[i] = ws + off; off += (size_t)N * nch[i];
  }
  float* lse0 = ws + off; off += N;
  float* lseT[4];
  for (int i = 1; i < 4; ++i) { lseT[i] = ws + off; off += N; }
  float* nll = ws + off; off += N;
  (void)ws_size; (void)in_sizes; (void)n_in; (void)out_size;

  const int BIG = 1 << 30;

  // 1) one-shot bf16 conversions
  auto cvt = [&](const float* s, unsigned short* d, size_t elems) {
    int np = (int)(elems / 2);
    cvt_bf16<<<dim3((np + 255) / 256), dim3(256), 0, stream>>>(s, (unsigned*)d, np);
  };
  cvt(hidden, hb, (size_t)N * DP);
  cvt(w0, w0b, (size_t)10000 * 1024);
  cvt(cw, cwb, (size_t)3 * 1024);
  cvt(w1, w1b, (size_t)10000 * 256);
  cvt(w2, w2b, (size_t)20000 * 64);
  pad_w3_bf16<<<dim3((10000 * 32 + 255) / 256), dim3(256), 0, stream>>>(w3, w3b);
  const float* ps[4] = {p0, p1, p2, p3};
  for (int i = 0; i < 4; ++i) {
    int elems = DP * dims[i];
    transpose_cvt_p<<<dim3((elems + 255) / 256), dim3(256), 0, stream>>>(ps[i], pTb[i], DP, dims[i]);
  }
  zero_pad_proj3b<<<dim3((4096 * 16 + 255) / 256), dim3(256), 0, stream>>>(projb[3]);

  // 2) proj_i = hidden @ pT_i^T  (f32 + bf16 dual store)
  float* projOut[4] = {proj0, proj1, proj2, proj3};
  const int pld[4] = {1024, 256, 64, 16};
  for (int i = 0; i < 4; ++i) {
    dim3 grid((dims[i] + 63) / 64, N / 64);
    gemm_bf16<<<grid, dim3(32), 0, stream>>>(hb, DP, DP,
                                             pTb[i], pTb[i], BIG,
                                             nullptr, nullptr, 0,
                                             dims[i], 0,
                                             projOut[i], pld[i], projb[i], pbld[i],
                                             nullptr, nullptr, 0);
  }

  // 3) head logsumexp: logits = proj0 @ concat(w0, cluster_w)^T + bias
  {
    dim3 grid(nch0, N / 64);
    gemm_bf16<<<grid, dim3(32), 0, stream>>>(projb[0], DP, DP,
                                             w0b, cwb, 10000,
                                             b0, cb, 10000,
                                             HEAD, 1,
                                             nullptr, 0, nullptr, 0, pmax0, psum0, nch0);
  }

  // 4) tail logsumexps (tail3 K padded to 32)
  const unsigned short* tW[4] = {nullptr, w1b, w2b, w3b};
  const float* tB[4] = {nullptr, b1, b2, b3};
  const int tK[4]    = {0, 256, 64, 32};
  const int tout[4]  = {0, 10000, 20000, 10000};
  for (int i = 1; i < 4; ++i) {
    dim3 grid(nch[i], N / 64);
    gemm_bf16<<<grid, dim3(32), 0, stream>>>(projb[i], tK[i], tK[i],
                                             tW[i], tW[i], BIG,
                                             tB[i], tB[i], BIG,
                                             tout[i], 1,
                                             nullptr, 0, nullptr, 0, pmaxT[i], psumT[i], nch[i]);
  }

  // 5) combine partials -> per-row LSE
  lse_combine<<<dim3((N + 255) / 256), dim3(256), 0, stream>>>(pmax0, psum0, lse0, nch0, N);
  for (int i = 1; i < 4; ++i)
    lse_combine<<<dim3((N + 255) / 256), dim3(256), 0, stream>>>(pmaxT[i], psumT[i], lseT[i], nch[i], N);

  // 6) exact-f32 gathered logits + per-token nll
  gather_nll<<<dim3(N), dim3(32), 0, stream>>>(proj0, proj1, proj2, proj3,
                                               w0, b0, w1, b1, w2, b2, w3, b3,
                                               cw, cb, target,
                                               lse0, lseT[1], lseT[2], lseT[3], nll);

  // 7) deterministic mean
  final_mean<<<dim3(1), dim3(256), 0, stream>>>(nll, out, N);
}
